// TriangleAttentionStartingNode_1279900254333
// MI455X (gfx1250) — matrix operand order repaired and checked by reading, not compiled, not run on hardware
//
#include <hip/hip_runtime.h>
#include <cstdint>

#define NSEQ 384
#define CZ 128
#define NH 4
#define HD 32

typedef __attribute__((ext_vector_type(16))) __bf16 v16bf;
typedef __attribute__((ext_vector_type(8))) float v8f;

union V16 { uint4 q[2]; v16bf bf; };

__device__ __forceinline__ unsigned short f2bf(float f) {
  unsigned int u = __float_as_uint(f);
  u = (u + 0x7FFFu + ((u >> 16) & 1u)) >> 16;  // round-to-nearest-even
  return (unsigned short)u;
}
__device__ __forceinline__ float bf2f(unsigned short h) {
  return __uint_as_float(((unsigned int)h) << 16);
}
__device__ __forceinline__ float redmax16(float v) {
#pragma unroll
  for (int m = 1; m < 16; m <<= 1) v = fmaxf(v, __shfl_xor(v, m, 16));
  return v;
}
__device__ __forceinline__ float redsum16(float v) {
#pragma unroll
  for (int m = 1; m < 16; m <<= 1) v += __shfl_xor(v, m, 16);
  return v;
}

// ---------------------------------------------------------------------------
// Kernel 0: convert 5 weight matrices (128x128 f32, [k][n]) to bf16 [n][k]
// so WMMA B-tiles become two contiguous b128 loads per lane.
// ---------------------------------------------------------------------------
__global__ __launch_bounds__(256) void k_prep(
    const float* __restrict__ Wq, const float* __restrict__ Wk,
    const float* __restrict__ Wv, const float* __restrict__ Wg,
    const float* __restrict__ Wo, unsigned short* __restrict__ WT) {
  int idx = blockIdx.x * 256 + threadIdx.x;            // 0 .. 81919
  int m = idx >> 14, e = idx & 16383;
  int n = e >> 7, k = e & 127;
  const float* src = (m == 0) ? Wq : (m == 1) ? Wk : (m == 2) ? Wv
                   : (m == 3) ? Wg : Wo;
  WT[(size_t)m * 16384 + (size_t)n * 128 + k] = f2bf(src[(size_t)k * 128 + n]);
}

// ---------------------------------------------------------------------------
// Kernel 1: LayerNorm + Q/K/V/G projections (bf16 WMMA) + bias = xln @ Wb.
// One wave = 16 rows. A-tile (16x128 bf16) staged in per-wave LDS.
// Stores: Qb/Kb/Gb as [i][h][j][d], Vt transposed as [i][h][d][k].
// ---------------------------------------------------------------------------
__global__ __launch_bounds__(256) void k_lnproj(
    const float* __restrict__ x, const float* __restrict__ gamma,
    const float* __restrict__ beta, const float* __restrict__ Wb,
    const float* __restrict__ bg, const unsigned short* __restrict__ WT,
    unsigned short* __restrict__ Qb, unsigned short* __restrict__ Kb,
    unsigned short* __restrict__ Vt, unsigned short* __restrict__ Gb,
    float* __restrict__ biasf) {
  __shared__ unsigned short a_tile[8][16 * 128];
  const int tid = threadIdx.x;
  const int wave = tid >> 5, lane = tid & 31, half = lane >> 4, lrow = lane & 15;
  const int i2 = blockIdx.x / 3;                        // first pair index
  const int jbase = (blockIdx.x % 3) * 128 + wave * 16; // second pair index base
  const size_t row = (size_t)blockIdx.x * 128 + wave * 16 + lrow;
  const int c0 = half * 64;
  const float* xr = x + row * CZ + c0;

  // --- LayerNorm statistics (two lanes per row, xor-16 combine) ---
  float s = 0.f, ss = 0.f;
#pragma unroll
  for (int f = 0; f < 64; f += 4) {
    float4 v = *(const float4*)(xr + f);
    s += v.x + v.y + v.z + v.w;
    ss += v.x * v.x + v.y * v.y + v.z * v.z + v.w * v.w;
  }
  s += __shfl_xor(s, 16, 32);
  ss += __shfl_xor(ss, 16, 32);
  const float mean = s * (1.f / 128.f);
  const float rstd = rsqrtf(ss * (1.f / 128.f) - mean * mean + 1e-5f);

  // --- normalize -> bf16 A-tile in LDS; fused bias = xln @ Wb partials ---
  float bp[4] = {0.f, 0.f, 0.f, 0.f};
#pragma unroll
  for (int f = 0; f < 64; f += 2) {
    float xa = xr[f], xb = xr[f + 1];
    float na = (xa - mean) * rstd * gamma[c0 + f] + beta[c0 + f];
    float nb = (xb - mean) * rstd * gamma[c0 + f + 1] + beta[c0 + f + 1];
    unsigned int pk = (unsigned int)f2bf(na) | ((unsigned int)f2bf(nb) << 16);
    *(unsigned int*)&a_tile[wave][lrow * 128 + c0 + f] = pk;
    float4 w0 = *(const float4*)(Wb + (size_t)(c0 + f) * 4);
    float4 w1 = *(const float4*)(Wb + (size_t)(c0 + f + 1) * 4);
    bp[0] += na * w0.x + nb * w1.x;
    bp[1] += na * w0.y + nb * w1.y;
    bp[2] += na * w0.z + nb * w1.z;
    bp[3] += na * w0.w + nb * w1.w;
  }
#pragma unroll
  for (int hh = 0; hh < 4; ++hh) bp[hh] += __shfl_xor(bp[hh], 16, 32);
  if (half == 0) {
    const int kb = jbase + lrow;
#pragma unroll
    for (int hh = 0; hh < 4; ++hh)
      biasf[((size_t)hh * NSEQ + i2) * NSEQ + kb] = bp[hh];  // [h][j][k]
  }
  asm volatile("s_wait_dscnt 0" ::: "memory");

  // --- 4 projections x 8 n-tiles x 4 K-chunks of v_wmma_f32_16x16x32_bf16 ---
#pragma unroll
  for (int m = 0; m < 4; ++m) {
    const unsigned short* W = WT + (size_t)m * 16384;  // Wq,Wk,Wv,Wg (bf16 [n][k])
    for (int nt = 0; nt < 8; ++nt) {
      v8f acc = {};
#pragma unroll
      for (int kc = 0; kc < 4; ++kc) {
        V16 a, b;
        const unsigned short* ap = &a_tile[wave][lrow * 128 + kc * 32 + half * 8];
        a.q[0] = *(const uint4*)ap;
        a.q[1] = *(const uint4*)(ap + 16);
        const unsigned short* bp_ =
            W + (size_t)(nt * 16 + lrow) * 128 + kc * 32 + half * 8;
        b.q[0] = *(const uint4*)bp_;
        b.q[1] = *(const uint4*)(bp_ + 16);
        acc = __builtin_amdgcn_wmma_f32_16x16x32_bf16(false, a.bf, false, b.bf,
                                                      (short)0, acc, false, false);
      }
      const int n = nt * 16 + lrow, h = n >> 5, d = n & 31;
      if (m == 3) {  // G: sigmoid(acc + bg) stored bf16
        const float bgv = bg[n];
#pragma unroll
        for (int rr = 0; rr < 8; ++rr) {
          const int j2 = jbase + rr + half * 8;
          const float v = acc[rr] + bgv;
          const float sg = 1.f / (1.f + __builtin_amdgcn_exp2f(-v * 1.4426950408889634f));
          Gb[(((size_t)i2 * NH + h) * NSEQ + j2) * HD + d] = f2bf(sg);
        }
      } else if (m == 2) {  // V stored transposed [i][h][d][k]
#pragma unroll
        for (int rr = 0; rr < 8; ++rr) {
          const int j2 = jbase + rr + half * 8;
          Vt[(((size_t)i2 * NH + h) * HD + d) * NSEQ + j2] = f2bf(acc[rr]);
        }
      } else {  // Q or K: [i][h][j][d]
        unsigned short* dst = (m == 0) ? Qb : Kb;
#pragma unroll
        for (int rr = 0; rr < 8; ++rr) {
          const int j2 = jbase + rr + half * 8;
          dst[(((size_t)i2 * NH + h) * NSEQ + j2) * HD + d] = f2bf(acc[rr]);
        }
      }
    }
  }
}

// ---------------------------------------------------------------------------
// Kernel 2: flash-style attention + gate + output projection.
// Grid (i, j-tile of 128); 8 waves x 16 rows each. Per head: QK^T is one
// WMMA per 16x16 tile (K = head_dim = 32), online softmax with xor-16
// reductions, P bounced through per-wave LDS (C->A layout), PV WMMAs,
// then (G*vals) @ Wo + bo via WMMA.
// ---------------------------------------------------------------------------
__global__ __launch_bounds__(256) void k_attn(
    const unsigned short* __restrict__ Qb, const unsigned short* __restrict__ Kb,
    const unsigned short* __restrict__ Vt, const unsigned short* __restrict__ Gb,
    const float* __restrict__ biasf, const float* __restrict__ mask,
    const unsigned short* __restrict__ WoT, const float* __restrict__ bo,
    float* __restrict__ out) {
  __shared__ unsigned short p_st[8][16 * 32];
  __shared__ unsigned short av[8][16 * 128];
  const int tid = threadIdx.x;
  const int wave = tid >> 5, lane = tid & 31, half = lane >> 4, lrow = lane & 15;
  const int i = blockIdx.x;
  const int j0 = blockIdx.y * 128 + wave * 16;
  const float scale = 0.17677669529663687f;  // 1/sqrt(32)
  const float L2E = 1.4426950408889634f;

  for (int h = 0; h < NH; ++h) {
    const size_t qbase = ((size_t)i * NH + h) * NSEQ;
    V16 aq;  // Q A-tile: 16 rows x 32 (=WMMA K) bf16
    {
      const unsigned short* qp = Qb + (qbase + (j0 + lrow)) * HD + half * 8;
      aq.q[0] = *(const uint4*)qp;
      aq.q[1] = *(const uint4*)(qp + 16);
    }
    float m8[8], l8[8];
    v8f oacc[2] = {};
#pragma unroll
    for (int rr = 0; rr < 8; ++rr) { m8[rr] = -1e30f; l8[rr] = 0.f; }
    const float* brow = biasf + ((size_t)h * NSEQ + j0) * NSEQ;

    for (int kc = 0; kc < NSEQ; kc += 32) {
      v8f s0 = {}, s1 = {};
      {
        V16 bk;  // B = K^T: lane = k column, contiguous d rows from Kb[i][h][k][d]
        const unsigned short* kp = Kb + (qbase + (kc + lrow)) * HD + half * 8;
        bk.q[0] = *(const uint4*)kp;
        bk.q[1] = *(const uint4*)(kp + 16);
        s0 = __builtin_amdgcn_wmma_f32_16x16x32_bf16(false, aq.bf, false, bk.bf,
                                                     (short)0, s0, false, false);
        const unsigned short* kp2 = Kb + (qbase + (kc + 16 + lrow)) * HD + half * 8;
        bk.q[0] = *(const uint4*)kp2;
        bk.q[1] = *(const uint4*)(kp2 + 16);
        s1 = __builtin_amdgcn_wmma_f32_16x16x32_bf16(false, aq.bf, false, bk.bf,
                                                     (short)0, s1, false, false);
      }
      const float mk0 = (1.f - mask[(size_t)i * NSEQ + kc + lrow]) * -1e9f;
      const float mk1 = (1.f - mask[(size_t)i * NSEQ + kc + 16 + lrow]) * -1e9f;
#pragma unroll
      for (int rr = 0; rr < 8; ++rr) {
        const int roff = (rr + half * 8) * NSEQ;
        float p0 = s0[rr] * scale + brow[roff + kc + lrow] + mk0;
        float p1 = s1[rr] * scale + brow[roff + kc + 16 + lrow] + mk1;
        const float t = redmax16(fmaxf(p0, p1));
        const float nm = fmaxf(m8[rr], t);
        const float a = __builtin_amdgcn_exp2f((m8[rr] - nm) * L2E);
        m8[rr] = nm;
        p0 = __builtin_amdgcn_exp2f((p0 - nm) * L2E);
        p1 = __builtin_amdgcn_exp2f((p1 - nm) * L2E);
        l8[rr] = l8[rr] * a + redsum16(p0 + p1);
        oacc[0][rr] *= a;
        oacc[1][rr] *= a;
        const int M = rr + half * 8;  // C-layout row -> A-layout staging
        p_st[wave][M * 32 + lrow] = f2bf(p0);
        p_st[wave][M * 32 + 16 + lrow] = f2bf(p1);
      }
      asm volatile("s_wait_dscnt 0" ::: "memory");
      V16 ap, bv;
      const unsigned short* pp = &p_st[wave][lrow * 32 + half * 8];
      ap.q[0] = *(const uint4*)pp;
      ap.q[1] = *(const uint4*)(pp + 16);
      const size_t vbase = ((size_t)i * NH + h) * HD;
      const unsigned short* vp0 = Vt + (vbase + lrow) * NSEQ + kc + half * 8;
      bv.q[0] = *(const uint4*)vp0;
      bv.q[1] = *(const uint4*)(vp0 + 16);
      oacc[0] = __builtin_amdgcn_wmma_f32_16x16x32_bf16(false, ap.bf, false, bv.bf,
                                                        (short)0, oacc[0], false, false);
      const unsigned short* vp1 = Vt + (vbase + 16 + lrow) * NSEQ + kc + half * 8;
      bv.q[0] = *(const uint4*)vp1;
      bv.q[1] = *(const uint4*)(vp1 + 16);
      oacc[1] = __builtin_amdgcn_wmma_f32_16x16x32_bf16(false, ap.bf, false, bv.bf,
                                                        (short)0, oacc[1], false, false);
    }
    // finalize head: divide by softmax sum, apply sigmoid gate, stage bf16
#pragma unroll
    for (int rr = 0; rr < 8; ++rr) {
      const float inv = 1.f / l8[rr];
      const int j = j0 + rr + half * 8;
#pragma unroll
      for (int dt = 0; dt < 2; ++dt) {
        const float gv = bf2f(Gb[(qbase + j) * HD + dt * 16 + lrow]);
        const float val = oacc[dt][rr] * inv * gv;
        av[wave][(rr + half * 8) * 128 + h * 32 + dt * 16 + lrow] = f2bf(val);
      }
    }
  }
  asm volatile("s_wait_dscnt 0" ::: "memory");

  // output projection: (G*vals)[16x128] @ Wo[128x128] + bo
  for (int nt = 0; nt < 8; ++nt) {
    v8f acc = {};
#pragma unroll
    for (int kc = 0; kc < 4; ++kc) {
      V16 a, b;
      const unsigned short* ap2 = &av[wave][lrow * 128 + kc * 32 + half * 8];
      a.q[0] = *(const uint4*)ap2;
      a.q[1] = *(const uint4*)(ap2 + 16);
      const unsigned short* bp2 =
          WoT + (size_t)(nt * 16 + lrow) * 128 + kc * 32 + half * 8;
      b.q[0] = *(const uint4*)bp2;
      b.q[1] = *(const uint4*)(bp2 + 16);
      acc = __builtin_amdgcn_wmma_f32_16x16x32_bf16(false, a.bf, false, b.bf,
                                                    (short)0, acc, false, false);
    }
    const float bov = bo[nt * 16 + lrow];
#pragma unroll
    for (int rr = 0; rr < 8; ++rr) {
      const int j = j0 + rr + half * 8;
      out[((size_t)i * NSEQ + j) * CZ + nt * 16 + lrow] = acc[rr] + bov;
    }
  }
}

// ---------------------------------------------------------------------------
extern "C" void kernel_launch(void* const* d_in, const int* in_sizes, int n_in,
                              void* d_out, int out_size, void* d_ws, size_t ws_size,
                              hipStream_t stream) {
  (void)in_sizes; (void)n_in; (void)out_size; (void)ws_size;
  const float* x     = (const float*)d_in[0];
  const float* mask  = (const float*)d_in[1];
  const float* gamma = (const float*)d_in[2];
  const float* beta  = (const float*)d_in[3];
  const float* Wq    = (const float*)d_in[4];
  const float* Wk    = (const float*)d_in[5];
  const float* Wv    = (const float*)d_in[6];
  const float* Wb    = (const float*)d_in[7];
  const float* Wg    = (const float*)d_in[8];
  const float* bg    = (const float*)d_in[9];
  const float* Wo    = (const float*)d_in[10];
  const float* bo    = (const float*)d_in[11];
  float* outF = (float*)d_out;

  char* ws = (char*)d_ws;
  unsigned short* WT = (unsigned short*)ws;                       // 5x128x128 bf16
  const size_t szW = (size_t)5 * 16384 * sizeof(unsigned short);  // 160 KB
  const size_t szQ = (size_t)NSEQ * NH * NSEQ * HD * sizeof(unsigned short);  // 36 MB
  unsigned short* Qb = (unsigned short*)(ws + szW);
  unsigned short* Kb = (unsigned short*)(ws + szW + szQ);
  unsigned short* Vt = (unsigned short*)(ws + szW + 2 * szQ);
  unsigned short* Gb = (unsigned short*)(ws + szW + 3 * szQ);
  float* biasf = (float*)(ws + szW + 4 * szQ);                    // [h][j][k] f32

  k_prep<<<320, 256, 0, stream>>>(Wq, Wk, Wv, Wg, Wo, WT);
  k_lnproj<<<1152, 256, 0, stream>>>(x, gamma, beta, Wb, bg, WT, Qb, Kb, Vt, Gb,
                                     biasf);
  k_attn<<<dim3(NSEQ, 3), 256, 0, stream>>>(Qb, Kb, Vt, Gb, biasf, mask,
                                            WT + 4 * 16384, bo, outF);
}
